// EmergenceModel_10299331575878
// MI455X (gfx1250) — matrix operand order repaired and checked by reading, not compiled, not run on hardware
//
#include <hip/hip_runtime.h>
#include <math.h>

typedef _Float16 h16;
typedef __attribute__((ext_vector_type(16))) _Float16 v16h;
typedef __attribute__((ext_vector_type(8)))  _Float16 v8h;
typedef __attribute__((ext_vector_type(8)))  float    v8f;

#define DD   1024
#define BB   8
#define SS   512
#define HH   8
#define HDIM 128
#define NTOK 4096

#define ACT_NONE 0
#define ACT_GELU 1
#define ACT_RELU 2

// ======================= device helpers =======================

__device__ __forceinline__ float gelu_erf(float x) {
  return 0.5f * x * (1.0f + erff(x * 0.70710678118654752f));
}
__device__ __forceinline__ float act_apply(float v, int act) {
  if (act == ACT_GELU) return gelu_erf(v);
  if (act == ACT_RELU) return fmaxf(v, 0.0f);
  return v;
}

// ======================= WMMA GEMM =======================
// C[M,Nn] = act(alpha * A[M,K] @ BT[Nn,K]^T + bias + resid)
// A row-major f16 (lda), BT row-major f16 (ldbt) == B^T, C f32 (ldc),
// optional dual f16 store C16 (ldc16; only non-batched callers use it).
// Wave tile: 32 rows x 64 cols (2 A-frags x 4 B-frags -> 8 WMMAs / K-step).
// Block = 4 waves stacked in M -> 128x64 block tile. All M in this graph are
// multiples of 128 (4096 tokens / 512 seq), so no partial M tiles.
// Fragment shapes per CDNA5 ISA 7.12.2 (16-bit A 16x32, B 32x16, f32 C 16x16); A/B k = 16*(lane/16)+s.
__global__ __launch_bounds__(128)
void k_gemm(const h16* __restrict__ A, const h16* __restrict__ BT,
            const float* __restrict__ bias, const float* __restrict__ resid,
            float* __restrict__ C, h16* __restrict__ C16,
            int M, int Nn, int K, int lda, int ldbt, int ldc, int ldc16,
            long sA, long sBT, long sC, float alpha, int act)
{
  const int lane = threadIdx.x & 31;
  const int wave = threadIdx.x >> 5;
  const int bz   = blockIdx.z;
  const h16* Ab = A  + (size_t)bz * sA;
  const h16* Bb = BT + (size_t)bz * sBT;
  const int row0 = blockIdx.y * 128 + wave * 32;
  const int n0   = blockIdx.x * 64;
  if (row0 >= M) return;

  v8f c00 = {}, c01 = {}, c02 = {}, c03 = {};
  v8f c10 = {}, c11 = {}, c12 = {}, c13 = {};

  // A: lane<16 -> row=lane, halves0..15=K+0..15
  //    lane>=16 -> same row, K offset +16
  const h16* ap0 = Ab + (size_t)(row0 + (lane & 15)) * lda + ((lane >> 4) << 4);
  const h16* ap1 = ap0 + (size_t)16 * lda;
  // B (as BT rows): lane col = lane&15, half j -> K = koff + j, koff = lane>=16 ? 16 : 0
  const h16* bp0 = Bb + (size_t)(n0 + (lane & 15)) * ldbt + ((lane >> 4) << 4);
  const size_t brow = (size_t)16 * ldbt;

  for (int k0 = 0; k0 < K; k0 += 32) {
    v16h a0, a1, b0, b1, b2, b3;
    *((v8h*)&a0)       = *((const v8h*)(ap0 + k0));
    *(((v8h*)&a0) + 1) = *((const v8h*)(ap0 + k0 + 8));
    *((v8h*)&a1)       = *((const v8h*)(ap1 + k0));
    *(((v8h*)&a1) + 1) = *((const v8h*)(ap1 + k0 + 8));
    const h16* bp = bp0 + k0;
    *((v8h*)&b0)       = *((const v8h*)(bp));
    *(((v8h*)&b0) + 1) = *((const v8h*)(bp + 8));
    bp += brow;
    *((v8h*)&b1)       = *((const v8h*)(bp));
    *(((v8h*)&b1) + 1) = *((const v8h*)(bp + 8));
    bp += brow;
    *((v8h*)&b2)       = *((const v8h*)(bp));
    *(((v8h*)&b2) + 1) = *((const v8h*)(bp + 8));
    bp += brow;
    *((v8h*)&b3)       = *((const v8h*)(bp));
    *(((v8h*)&b3) + 1) = *((const v8h*)(bp + 8));
    __builtin_prefetch(ap0 + k0 + 128, 0, 3);   // global_prefetch_b8, near-cache hint
    __builtin_prefetch(ap1 + k0 + 128, 0, 3);
    c00 = __builtin_amdgcn_wmma_f32_16x16x32_f16(false, a0, false, b0, (short)0, c00, false, false);
    c01 = __builtin_amdgcn_wmma_f32_16x16x32_f16(false, a0, false, b1, (short)0, c01, false, false);
    c02 = __builtin_amdgcn_wmma_f32_16x16x32_f16(false, a0, false, b2, (short)0, c02, false, false);
    c03 = __builtin_amdgcn_wmma_f32_16x16x32_f16(false, a0, false, b3, (short)0, c03, false, false);
    c10 = __builtin_amdgcn_wmma_f32_16x16x32_f16(false, a1, false, b0, (short)0, c10, false, false);
    c11 = __builtin_amdgcn_wmma_f32_16x16x32_f16(false, a1, false, b1, (short)0, c11, false, false);
    c12 = __builtin_amdgcn_wmma_f32_16x16x32_f16(false, a1, false, b2, (short)0, c12, false, false);
    c13 = __builtin_amdgcn_wmma_f32_16x16x32_f16(false, a1, false, b3, (short)0, c13, false, false);
  }

  // C/D layout: VGPR r -> row = rowbase + r + (lane>=16 ? 8 : 0), col = n0 + (lane&15)
  const int rb0 = row0 + ((lane >> 4) << 3);
  const int cb  = n0 + (lane & 15);
  float* Cb = C ? (C + (size_t)bz * sC) : (float*)0;

  auto store_tile = [&](const v8f& acc, int rowoff, int tt) {
    const int col = cb + 16 * tt;
    if (col >= Nn) return;
    const float bv = bias ? bias[col] : 0.0f;
#pragma unroll
    for (int r = 0; r < 8; ++r) {
      const int row = rb0 + rowoff + r;
      float v = acc[r] * alpha + bv;
      if (resid) v += resid[(size_t)bz * sC + (size_t)row * ldc + col];
      v = act_apply(v, act);
      if (Cb)  Cb[(size_t)row * ldc + col] = v;
      if (C16) C16[(size_t)row * ldc16 + col] = (h16)v;
    }
  };
  store_tile(c00,  0, 0); store_tile(c01,  0, 1); store_tile(c02,  0, 2); store_tile(c03,  0, 3);
  store_tile(c10, 16, 0); store_tile(c11, 16, 1); store_tile(c12, 16, 2); store_tile(c13, 16, 3);
}

// ======================= conversion / reshape kernels =======================

// dst[r*ldd + c0 + c] = (c<cols) ? src[r*cols+c] : 0   (region width wcols)
__global__ void k_cvt(const float* __restrict__ src, h16* __restrict__ dst,
                      int rows, int cols, int wcols, int ldd, int c0)
{
  size_t i = (size_t)blockIdx.x * blockDim.x + threadIdx.x;
  size_t tot = (size_t)rows * wcols;
  if (i >= tot) return;
  int c = (int)(i % wcols);
  size_t r = i / wcols;
  float v = (c < cols) ? src[r * cols + c] : 0.0f;
  dst[r * ldd + c0 + c] = (h16)v;
}

// W[K,N] f32 -> WT[Npad,K] f16 (zero padded rows)
__global__ void k_cvtT(const float* __restrict__ w, h16* __restrict__ dst,
                       int K, int N, int Npad)
{
  size_t i = (size_t)blockIdx.x * blockDim.x + threadIdx.x;
  size_t tot = (size_t)Npad * K;
  if (i >= tot) return;
  int k = (int)(i % K);
  int n = (int)(i / K);
  dst[i] = (n < N) ? (h16)w[(size_t)k * N + n] : (h16)0.0f;
}

// x[Ntok, ldsrc] (+c0) -> dst f16 [B,H,S,hd]
__global__ void k_heads(const float* __restrict__ x, h16* __restrict__ dst, int ldsrc, int c0)
{
  size_t i = (size_t)blockIdx.x * blockDim.x + threadIdx.x;
  if (i >= (size_t)NTOK * DD) return;
  int d = (int)(i % HDIM);
  size_t q = i / HDIM;
  int s = (int)(q % SS); q /= SS;
  int h = (int)(q % HH);
  int b = (int)(q / HH);
  int token = b * SS + s;
  dst[i] = (h16)x[(size_t)token * ldsrc + c0 + h * HDIM + d];
}

// x[Ntok, ldsrc] (+c0) -> dst f16 [B,H,hd,S]  (v transposed per head)
__global__ void k_headsT(const float* __restrict__ x, h16* __restrict__ dst, int ldsrc, int c0)
{
  size_t i = (size_t)blockIdx.x * blockDim.x + threadIdx.x;
  if (i >= (size_t)NTOK * DD) return;
  int s = (int)(i % SS);
  size_t q = i / SS;
  int d = (int)(q % HDIM); q /= HDIM;
  int h = (int)(q % HH);
  int b = (int)(q / HH);
  int token = b * SS + s;
  dst[i] = (h16)x[(size_t)token * ldsrc + c0 + h * HDIM + d];
}

// o f32 [B,H,S,hd] -> dst f16 [Ntok, D]
__global__ void k_merge(const float* __restrict__ o, h16* __restrict__ dst)
{
  size_t i = (size_t)blockIdx.x * blockDim.x + threadIdx.x;
  if (i >= (size_t)NTOK * DD) return;
  int col = (int)(i % DD);
  int token = (int)(i / DD);
  int h = col / HDIM, d = col % HDIM;
  int b = token / SS, s = token % SS;
  dst[i] = (h16)o[(((size_t)(b * HH + h) * SS) + s) * HDIM + d];
}

// ======================= row kernels =======================

__global__ __launch_bounds__(256)
void k_softmax(const float* __restrict__ sc, h16* __restrict__ pr, int L)
{
  __shared__ float red[256];
  __shared__ float s_max, s_sum;
  const size_t row = blockIdx.x;
  const float* x = sc + row * L;
  h16* p = pr + row * L;
  float m = -3.4e38f;
  for (int j = threadIdx.x; j < L; j += 256) m = fmaxf(m, x[j]);
  red[threadIdx.x] = m; __syncthreads();
  for (int o = 128; o > 0; o >>= 1) { if (threadIdx.x < o) red[threadIdx.x] = fmaxf(red[threadIdx.x], red[threadIdx.x + o]); __syncthreads(); }
  if (threadIdx.x == 0) s_max = red[0];
  __syncthreads();
  const float mm = s_max;
  float sum = 0.0f;
  for (int j = threadIdx.x; j < L; j += 256) sum += expf(x[j] - mm);
  red[threadIdx.x] = sum; __syncthreads();
  for (int o = 128; o > 0; o >>= 1) { if (threadIdx.x < o) red[threadIdx.x] += red[threadIdx.x + o]; __syncthreads(); }
  if (threadIdx.x == 0) s_sum = red[0];
  __syncthreads();
  const float inv = 1.0f / s_sum;
  for (int j = threadIdx.x; j < L; j += 256) p[j] = (h16)(expf(x[j] - mm) * inv);
}

__global__ __launch_bounds__(256)
void k_sqnorm(const float* __restrict__ f, float* __restrict__ sq, int d)
{
  __shared__ float red[256];
  const size_t row = blockIdx.x;
  float s = 0.0f;
  for (int j = threadIdx.x; j < d; j += 256) { float v = f[row * d + j]; s += v * v; }
  red[threadIdx.x] = s; __syncthreads();
  for (int o = 128; o > 0; o >>= 1) { if (threadIdx.x < o) red[threadIdx.x] += red[threadIdx.x + o]; __syncthreads(); }
  if (threadIdx.x == 0) sq[row] = red[0];
}

// EntropyController row: kern row staged in LDS (n*4 bytes dynamic)
__global__ __launch_bounds__(256)
void k_ecrow(const float* __restrict__ gram, const float* __restrict__ sq,
             const float* __restrict__ f, float* __restrict__ fout,
             float* __restrict__ ctrl, const float* __restrict__ target,
             const float* __restrict__ temp, int n, int d)
{
  extern __shared__ float kern[];
  __shared__ float red[256];
  __shared__ float s_sum, s_ctrl;
  const int i = blockIdx.x;
  const float tp = temp[0];
  const float inv2t2 = 1.0f / (2.0f * tp * tp);
  const float sqi = sq[i];
  const float* grow = gram + (size_t)i * n;
  float lsum = 0.0f;
  for (int j = threadIdx.x; j < n; j += 256) {
    float d2 = fmaxf(sqi + sq[j] - 2.0f * grow[j], 0.0f);
    float kv = expf(-d2 * inv2t2);
    kern[j] = kv; lsum += kv;
  }
  red[threadIdx.x] = lsum; __syncthreads();
  for (int o = 128; o > 0; o >>= 1) { if (threadIdx.x < o) red[threadIdx.x] += red[threadIdx.x + o]; __syncthreads(); }
  if (threadIdx.x == 0) s_sum = red[0];
  __syncthreads();
  const float inv = 1.0f / s_sum;
  float lent = 0.0f;
  for (int j = threadIdx.x; j < n; j += 256) { float pj = kern[j] * inv; lent -= pj * logf(pj + 1e-6f); }
  red[threadIdx.x] = lent; __syncthreads();
  for (int o = 128; o > 0; o >>= 1) { if (threadIdx.x < o) red[threadIdx.x] += red[threadIdx.x + o]; __syncthreads(); }
  if (threadIdx.x == 0) {
    float ent = red[0];
    float cc = 1.0f / (1.0f + expf((ent - target[0]) / tp));  // sigmoid(-(ent-target)/t)
    s_ctrl = cc; ctrl[i] = cc;
  }
  __syncthreads();
  const float cc = s_ctrl;
  for (int j = threadIdx.x; j < d; j += 256) fout[(size_t)i * d + j] = f[(size_t)i * d + j] * cc;
}

// LayerNorm row (+ optional act), f32 and/or f16 out
__global__ __launch_bounds__(256)
void k_ln(const float* __restrict__ x, const float* __restrict__ g, const float* __restrict__ b,
          float* __restrict__ y, h16* __restrict__ y16, int d, int act)
{
  __shared__ float red[256];
  __shared__ float s_mu, s_rstd;
  const size_t row = blockIdx.x;
  const float* xr = x + row * d;
  float s = 0.0f;
  for (int j = threadIdx.x; j < d; j += 256) s += xr[j];
  red[threadIdx.x] = s; __syncthreads();
  for (int o = 128; o > 0; o >>= 1) { if (threadIdx.x < o) red[threadIdx.x] += red[threadIdx.x + o]; __syncthreads(); }
  if (threadIdx.x == 0) s_mu = red[0] / d;
  __syncthreads();
  const float mu = s_mu;
  float v = 0.0f;
  for (int j = threadIdx.x; j < d; j += 256) { float dv = xr[j] - mu; v += dv * dv; }
  red[threadIdx.x] = v; __syncthreads();
  for (int o = 128; o > 0; o >>= 1) { if (threadIdx.x < o) red[threadIdx.x] += red[threadIdx.x + o]; __syncthreads(); }
  if (threadIdx.x == 0) s_rstd = rsqrtf(red[0] / d + 1e-5f);
  __syncthreads();
  const float rstd = s_rstd;
  for (int j = threadIdx.x; j < d; j += 256) {
    float o = (xr[j] - mu) * rstd * g[j] + b[j];
    o = act_apply(o, act);
    if (y)   y[row * d + j] = o;
    if (y16) y16[row * d + j] = (h16)o;
  }
}

__global__ void k_rowscale(const float* __restrict__ x, const float* __restrict__ s,
                           float* __restrict__ y, h16* __restrict__ y16, int d)
{
  size_t i = (size_t)blockIdx.x * blockDim.x + threadIdx.x;
  if (i >= (size_t)NTOK * d) return;
  size_t r = i / d;
  float v = x[i] * s[r];
  if (y)   y[i] = v;
  if (y16) y16[i] = (h16)v;
}

__global__ void k_phase(const float* __restrict__ e, const float* __restrict__ temp,
                        float* __restrict__ phase, int n)
{
  int i = blockIdx.x * blockDim.x + threadIdx.x;
  if (i >= n) return;
  phase[i] = 1.0f / (1.0f + expf(-e[i] / temp[0]));
}

__global__ void k_avg2(const float* __restrict__ a, const float* __restrict__ b,
                       float* __restrict__ y, int n)
{
  int i = blockIdx.x * blockDim.x + threadIdx.x;
  if (i >= n) return;
  y[i] = 0.5f * (a[i] + b[i]);
}

// ======================= host orchestration =======================

struct Arena {
  char* base; size_t off; size_t cap;
  size_t stk[32]; int sp;
  void* alloc(size_t bytes) {
    size_t a = (off + 255) & ~(size_t)255;
    off = a + bytes;
    return base + a;
  }
  void push() { stk[sp++] = off; }
  void pop()  { off = stk[--sp]; }
};

struct Ctx {
  hipStream_t st;
  Arena ar;
  void* const* din;
  const float* P(int i) const { return (const float*)din[i]; }
};

static inline int blks(size_t tot) { return (int)((tot + 255) / 256); }

static h16* cvtW(Ctx& c, int widx, int K, int N) {
  int Npad = (N + 63) & ~63;
  h16* dst = (h16*)c.ar.alloc((size_t)Npad * K * sizeof(h16));
  size_t tot = (size_t)Npad * K;
  k_cvtT<<<blks(tot), 256, 0, c.st>>>(c.P(widx), dst, K, N, Npad);
  return dst;
}

static void cvt_into(Ctx& c, const float* src, h16* dst, int rows, int cols, int ldd, int c0) {
  size_t tot = (size_t)rows * cols;
  k_cvt<<<blks(tot), 256, 0, c.st>>>(src, dst, rows, cols, cols, ldd, c0);
}

static h16* cvtA(Ctx& c, const float* x, int M, int K) {
  h16* dst = (h16*)c.ar.alloc((size_t)M * K * sizeof(h16));
  cvt_into(c, x, dst, M, K, K, 0);
  return dst;
}

static void gemm(Ctx& c, const h16* A, const h16* BT, const float* bias, const float* resid,
                 float* C, h16* C16, int M, int Nn, int K,
                 int lda, int ldbt, int ldc, int ldc16,
                 float alpha, int act, int batch = 1, long sA = 0, long sBT = 0, long sC = 0)
{
  dim3 grid((Nn + 63) / 64, (M + 127) / 128, batch);
  k_gemm<<<grid, 128, 0, c.st>>>(A, BT, bias, resid, C, C16, M, Nn, K,
                                 lda, ldbt, ldc, ldc16, sA, sBT, sC, alpha, act);
}

// Y = act(X16 @ W(widx) + b(bidx) [+ resid]); X16 is [M,K] ld=K
static void linear(Ctx& c, const h16* X16, int M, int K, int widx, int bidx, int Nn,
                   float* Y, h16* Y16, int act, const float* resid = nullptr)
{
  c.ar.push();
  h16* WT = cvtW(c, widx, K, Nn);
  gemm(c, X16, WT, c.P(bidx), resid, Y, Y16, M, Nn, K, K, K, Nn, Nn, 1.0f, act);
  c.ar.pop();
}

// qh[B,H,S,hd] kh[B,H,S,hd] vT[B,H,hd,S] (f16) -> merged16 [Ntok,D] f16
static void attention(Ctx& c, const h16* qh, const h16* kh, const h16* vT, h16* merged16)
{
  c.ar.push();
  const size_t sc_elems = (size_t)BB * HH * SS * SS;
  float* scores = (float*)c.ar.alloc(sc_elems * sizeof(float));
  h16*   probs  = (h16*)c.ar.alloc(sc_elems * sizeof(h16));
  float* o      = (float*)c.ar.alloc((size_t)NTOK * DD * sizeof(float));
  gemm(c, qh, kh, nullptr, nullptr, scores, nullptr, SS, SS, HDIM,
       HDIM, HDIM, SS, 0, 0.088388347648318447f /*1/sqrt(128)*/, ACT_NONE,
       BB * HH, (long)SS * HDIM, (long)SS * HDIM, (long)SS * SS);
  k_softmax<<<BB * HH * SS, 256, 0, c.st>>>(scores, probs, SS);
  gemm(c, probs, vT, nullptr, nullptr, o, nullptr, SS, HDIM, SS,
       SS, SS, HDIM, 0, 1.0f, ACT_NONE,
       BB * HH, (long)SS * SS, (long)HDIM * SS, (long)SS * HDIM);
  k_merge<<<blks((size_t)NTOK * DD), 256, 0, c.st>>>(o, merged16);
  c.ar.pop();
}

struct MhaIdx { int bo, bqkv, wo, wqkv; };
static void mha(Ctx& c, const h16* x16, MhaIdx id, float* out, h16* out16)
{
  c.ar.push();
  float* qkv = (float*)c.ar.alloc((size_t)NTOK * 3 * DD * sizeof(float));
  {
    c.ar.push();
    h16* W = cvtW(c, id.wqkv, DD, 3 * DD);
    gemm(c, x16, W, c.P(id.bqkv), nullptr, qkv, nullptr, NTOK, 3 * DD, DD,
         DD, DD, 3 * DD, 0, 1.0f, ACT_NONE);
    c.ar.pop();
  }
  h16* qh = (h16*)c.ar.alloc((size_t)NTOK * DD * sizeof(h16));
  h16* kh = (h16*)c.ar.alloc((size_t)NTOK * DD * sizeof(h16));
  h16* vT = (h16*)c.ar.alloc((size_t)NTOK * DD * sizeof(h16));
  h16* mg = (h16*)c.ar.alloc((size_t)NTOK * DD * sizeof(h16));
  const size_t tot = (size_t)NTOK * DD;
  k_heads <<<blks(tot), 256, 0, c.st>>>(qkv, qh, 3 * DD, 0);
  k_heads <<<blks(tot), 256, 0, c.st>>>(qkv, kh, 3 * DD, DD);
  k_headsT<<<blks(tot), 256, 0, c.st>>>(qkv, vT, 3 * DD, 2 * DD);
  attention(c, qh, kh, vT, mg);
  linear(c, mg, NTOK, DD, id.wo, id.bo, DD, out, out16, ACT_NONE);
  c.ar.pop();
}

struct CmaIdx { int kb, kw, ob, ow, qb, qw, vb, vw; };
static void cma(Ctx& c, const h16* qf16, const float* qresid, const h16* kv16, CmaIdx id,
                float* out, h16* out16)
{
  c.ar.push();
  h16* qh = (h16*)c.ar.alloc((size_t)NTOK * DD * sizeof(h16));
  h16* kh = (h16*)c.ar.alloc((size_t)NTOK * DD * sizeof(h16));
  h16* vT = (h16*)c.ar.alloc((size_t)NTOK * DD * sizeof(h16));
  h16* mg = (h16*)c.ar.alloc((size_t)NTOK * DD * sizeof(h16));
  float* tmp = (float*)c.ar.alloc((size_t)NTOK * DD * sizeof(float));
  const size_t tot = (size_t)NTOK * DD;
  linear(c, qf16, NTOK, DD, id.qw, id.qb, DD, tmp, nullptr, ACT_NONE);
  k_heads <<<blks(tot), 256, 0, c.st>>>(tmp, qh, DD, 0);
  linear(c, kv16, NTOK, DD, id.kw, id.kb, DD, tmp, nullptr, ACT_NONE);
  k_heads <<<blks(tot), 256, 0, c.st>>>(tmp, kh, DD, 0);
  linear(c, kv16, NTOK, DD, id.vw, id.vb, DD, tmp, nullptr, ACT_NONE);
  k_headsT<<<blks(tot), 256, 0, c.st>>>(tmp, vT, DD, 0);
  attention(c, qh, kh, vT, mg);
  linear(c, mg, NTOK, DD, id.ow, id.ob, DD, out, out16, ACT_NONE, qresid);
  c.ar.pop();
}

struct CtIdx { int m1b, m1w, m2b, m2w, m3b, m3w; MhaIdx mh; };
static void ct(Ctx& c, const h16* x16, const float* xresid, CtIdx id, float* out, h16* out16)
{
  c.ar.push();
  h16* a16 = (h16*)c.ar.alloc((size_t)NTOK * DD * sizeof(h16));
  mha(c, x16, id.mh, nullptr, a16);
  h16* h1 = (h16*)c.ar.alloc((size_t)NTOK * 4 * DD * sizeof(h16));
  linear(c, a16, NTOK, DD, id.m1w, id.m1b, 4 * DD, nullptr, h1, ACT_GELU);
  h16* h2 = (h16*)c.ar.alloc((size_t)NTOK * 2 * DD * sizeof(h16));
  linear(c, h1, NTOK, 4 * DD, id.m2w, id.m2b, 2 * DD, nullptr, h2, ACT_GELU);
  linear(c, h2, NTOK, 2 * DD, id.m3w, id.m3b, DD, out, out16, ACT_NONE, xresid);
  c.ar.pop();
}

struct EpIdx { int p1b, p1w, p2b, p2w; };
static void epred(Ctx& c, const h16* x16, EpIdx id, float* out, h16* out16)
{
  c.ar.push();
  h16* h1 = (h16*)c.ar.alloc((size_t)NTOK * 2 * DD * sizeof(h16));
  linear(c, x16, NTOK, DD, id.p1w, id.p1b, 2 * DD, nullptr, h1, ACT_RELU);
  linear(c, h1, NTOK, 2 * DD, id.p2w, id.p2b, DD, out, out16, ACT_NONE);
  c.ar.pop();
}

static void ec(Ctx& c, const float* f, const h16* f16, int tidx, int tempidx,
               float* fout, float* ctrl)
{
  c.ar.push();
  float* gram = (float*)c.ar.alloc((size_t)NTOK * NTOK * sizeof(float));
  gemm(c, f16, f16, nullptr, nullptr, gram, nullptr, NTOK, NTOK, DD,
       DD, DD, NTOK, 0, 1.0f, ACT_NONE);
  float* sq = (float*)c.ar.alloc(NTOK * sizeof(float));
  k_sqnorm<<<NTOK, 256, 0, c.st>>>(f, sq, DD);
  k_ecrow<<<NTOK, 256, NTOK * sizeof(float), c.st>>>(
      gram, sq, f, fout, ctrl, c.P(tidx), c.P(tempidx), NTOK, DD);
  c.ar.pop();
}

struct PtIdx { int e1b, e1w, e2b, e2w, e3b, e3w, ect, ectemp, lnb, lng, temp; };
static void pt(Ctx& c, const h16* j16, const float* t, const h16* t16,
               const float* i_, const h16* i16, PtIdx id,
               float* phase, float* tc, float* ic, float* ew)
{
  c.ar.push();
  h16* g1 = (h16*)c.ar.alloc((size_t)NTOK * 4 * DD * sizeof(h16));
  linear(c, j16, NTOK, 2 * DD, id.e1w, id.e1b, 4 * DD, nullptr, g1, ACT_GELU);
  float* h2 = (float*)c.ar.alloc((size_t)NTOK * 2 * DD * sizeof(float));
  linear(c, g1, NTOK, 4 * DD, id.e2w, id.e2b, 2 * DD, h2, nullptr, ACT_NONE);
  h16* h3 = (h16*)c.ar.alloc((size_t)NTOK * 2 * DD * sizeof(h16));
  k_ln<<<NTOK, 256, 0, c.st>>>(h2, c.P(id.lng), c.P(id.lnb), nullptr, h3, 2 * DD, ACT_GELU);
  float* e = (float*)c.ar.alloc(NTOK * sizeof(float));
  linear(c, h3, NTOK, 2 * DD, id.e3w, id.e3b, 1, e, nullptr, ACT_NONE);
  k_phase<<<blks(NTOK), 256, 0, c.st>>>(e, c.P(id.temp), phase, NTOK);
  float* tw = (float*)c.ar.alloc(NTOK * sizeof(float));
  float* iw = (float*)c.ar.alloc(NTOK * sizeof(float));
  ec(c, t,  t16, id.ect, id.ectemp, tc, tw);
  ec(c, i_, i16, id.ect, id.ectemp, ic, iw);
  if (ew) k_avg2<<<blks(NTOK), 256, 0, c.st>>>(tw, iw, ew, NTOK);
  c.ar.pop();
}

struct EcoreIdx { CmaIdx cm; CtIdx ctx; EpIdx ep; PtIdx ptx; };
static void ecore(Ctx& c, const float* x, const h16* x16, const float* ctxf, const h16* ctx16,
                  const h16* j16, EcoreIdx id, float* out, h16* out16)
{
  c.ar.push();
  float* phase = (float*)c.ar.alloc(NTOK * sizeof(float));
  float* xp = (float*)c.ar.alloc((size_t)NTOK * DD * sizeof(float));
  float* cp = (float*)c.ar.alloc((size_t)NTOK * DD * sizeof(float));
  pt(c, j16, x, x16, ctxf, ctx16, id.ptx, phase, xp, cp, nullptr);
  float* xs = (float*)c.ar.alloc((size_t)NTOK * DD * sizeof(float));
  h16* xs16 = (h16*)c.ar.alloc((size_t)NTOK * DD * sizeof(h16));
  k_rowscale<<<blks((size_t)NTOK * DD), 256, 0, c.st>>>(xp, phase, xs, xs16, DD);
  float* cs = (float*)c.ar.alloc((size_t)NTOK * DD * sizeof(float));
  h16* cs16 = (h16*)c.ar.alloc((size_t)NTOK * DD * sizeof(h16));
  ct(c, xs16, xs, id.ctx, cs, cs16);
  h16* cp16 = cvtA(c, cp, NTOK, DD);
  float* cs2 = (float*)c.ar.alloc((size_t)NTOK * DD * sizeof(float));
  h16* cs2_16 = (h16*)c.ar.alloc((size_t)NTOK * DD * sizeof(h16));
  cma(c, cs16, cs, cp16, id.cm, cs2, cs2_16);
  epred(c, cs2_16, id.ep, out, out16);
  c.ar.pop();
}

// ---- flattened param indices (jax pytree: sorted dict keys; d_in[0]=text, d_in[1]=image) ----
extern "C" void kernel_launch(void* const* d_in, const int* in_sizes, int n_in,
                              void* d_out, int out_size, void* d_ws, size_t ws_size,
                              hipStream_t stream)
{
  (void)in_sizes; (void)n_in; (void)out_size;
  Ctx c; c.st = stream; c.din = d_in;
  c.ar.base = (char*)d_ws; c.ar.off = 0; c.ar.cap = ws_size; c.ar.sp = 0;

  const CmaIdx CM_BI    = {2, 3, 4, 5, 6, 7, 8, 9};
  const int F1B = 10, F1W = 11, F2B = 12, F2W = 13;
  const EcoreIdx IE = { {14,15,16,17,18,19,20,21},
                        {22,23,24,25,26,27, {28,29,30,31}},
                        {32,33,34,35},
                        {36,37,38,39,40,41, 42,43, 44,45, 46} };
  const EcoreIdx TE = { {47,48,49,50,51,52,53,54},
                        {55,56,57,58,59,60, {61,62,63,64}},
                        {65,66,67,68},
                        {69,70,71,72,73,74, 75,76, 77,78, 79} };
  const int M2MB = 80, M2MW = 81;
  const CtIdx CT_MACRO = {82,83,84,85,86,87, {88,89,90,91}};
  const EpIdx EP_MACRO = {92,93,94,95};
  const int MESO_ECT = 96, MESO_ECTEMP = 97;
  const int G1B = 98, G1W = 99, G2B = 100, G2W = 101, GLNB = 102, GLNG = 103;
  const MhaIdx MHA_MESO = {104,105,106,107};
  const CmaIdx CA_MICRO = {108,109,110,111,112,113,114,115};
  const PtIdx PT_MICRO = {116,117,118,119,120,121, 122,123, 124,125, 126};
  const int SIB = 127, SIW = 128;

  const float* T = (const float*)d_in[0];
  const float* I = (const float*)d_in[1];
  float* out = (float*)d_out;
  float* out_ft = out;                                   // [N, 3D]
  float* out_fi = out + (size_t)NTOK * 3 * DD;           // [N, 3D]
  float* out_ge = out + 2 * (size_t)NTOK * 3 * DD;       // [N, D]
  float* out_ew = out_ge + (size_t)NTOK * DD;            // [N]
  float* out_sg = out_ew + NTOK;                         // [N, D]

  // persistent stagings
  h16* T16 = cvtA(c, T, NTOK, DD);
  h16* I16 = cvtA(c, I, NTOK, DD);
  h16* jTI = (h16*)c.ar.alloc((size_t)NTOK * 2 * DD * sizeof(h16));
  cvt_into(c, T, jTI, NTOK, DD, 2 * DD, 0);
  cvt_into(c, I, jTI, NTOK, DD, 2 * DD, DD);
  h16* jIT = (h16*)c.ar.alloc((size_t)NTOK * 2 * DD * sizeof(h16));
  cvt_into(c, I, jIT, NTOK, DD, 2 * DD, 0);
  cvt_into(c, T, jIT, NTOK, DD, 2 * DD, DD);

  float* ewM = (float*)c.ar.alloc(NTOK * sizeof(float));
  float* ewS = (float*)c.ar.alloc(NTOK * sizeof(float));
  float* local = (float*)c.ar.alloc((size_t)NTOK * DD * sizeof(float));
  h16* local16 = (h16*)c.ar.alloc((size_t)NTOK * DD * sizeof(h16));
  h16* sg16 = (h16*)c.ar.alloc((size_t)NTOK * DD * sizeof(h16));

  // ---- micro: LocalFeatureAligner ----
  {
    c.ar.push();
    float* phaseM = (float*)c.ar.alloc(NTOK * sizeof(float));
    float* tp = (float*)c.ar.alloc((size_t)NTOK * DD * sizeof(float));
    float* ip = (float*)c.ar.alloc((size_t)NTOK * DD * sizeof(float));
    pt(c, jTI, T, T16, I, I16, PT_MICRO, phaseM, tp, ip, ewM);
    h16* tp16 = cvtA(c, tp, NTOK, DD);
    h16* ip16 = cvtA(c, ip, NTOK, DD);
    float* cm = (float*)c.ar.alloc((size_t)NTOK * DD * sizeof(float));
    cma(c, tp16, tp, ip16, CA_MICRO, cm, nullptr);
    float* lp = (float*)c.ar.alloc((size_t)NTOK * DD * sizeof(float));
    h16* lp16 = (h16*)c.ar.alloc((size_t)NTOK * DD * sizeof(h16));
    k_rowscale<<<blks((size_t)NTOK * DD), 256, 0, c.st>>>(cm, phaseM, lp, lp16, DD);
    linear(c, lp16, NTOK, DD, M2MW, M2MB, DD, local, local16, ACT_NONE);
    c.ar.pop();
  }

  // ---- meso: SemanticGraphBuilder ----
  {
    c.ar.push();
    float* att = (float*)c.ar.alloc((size_t)NTOK * DD * sizeof(float));
    h16* att16 = (h16*)c.ar.alloc((size_t)NTOK * DD * sizeof(h16));
    mha(c, local16, MHA_MESO, att, att16);
    float* atts = (float*)c.ar.alloc((size_t)NTOK * DD * sizeof(float));
    ec(c, att, att16, MESO_ECT, MESO_ECTEMP, atts, ewS);
    h16* atts16 = cvtA(c, atts, NTOK, DD);
    float* g1f = (float*)c.ar.alloc((size_t)NTOK * 2 * DD * sizeof(float));
    linear(c, atts16, NTOK, DD, G1W, G1B, 2 * DD, g1f, nullptr, ACT_NONE);
    h16* gh16 = (h16*)c.ar.alloc((size_t)NTOK * 2 * DD * sizeof(h16));
    k_ln<<<NTOK, 256, 0, c.st>>>(g1f, c.P(GLNG), c.P(GLNB), nullptr, gh16, 2 * DD, ACT_GELU);
    linear(c, gh16, NTOK, 2 * DD, G2W, G2B, DD, out_sg, sg16, ACT_NONE);
    c.ar.pop();
  }

  // ew = ((tw+iw)/2 + meso)/2
  k_avg2<<<blks(NTOK), 256, 0, c.st>>>(ewM, ewS, out_ew, NTOK);

  // ---- macro: GlobalEmergenceLayer ----
  {
    c.ar.push();
    float* ctm = (float*)c.ar.alloc((size_t)NTOK * DD * sizeof(float));
    h16* ctm16 = (h16*)c.ar.alloc((size_t)NTOK * DD * sizeof(h16));
    ct(c, sg16, out_sg, CT_MACRO, ctm, ctm16);
    epred(c, ctm16, EP_MACRO, out_ge, nullptr);
    c.ar.pop();
  }

  // ---- bidirectional emergence ----
  float* te = (float*)c.ar.alloc((size_t)NTOK * DD * sizeof(float));
  h16* te16 = (h16*)c.ar.alloc((size_t)NTOK * DD * sizeof(h16));
  float* ie = (float*)c.ar.alloc((size_t)NTOK * DD * sizeof(float));
  h16* ie16 = (h16*)c.ar.alloc((size_t)NTOK * DD * sizeof(h16));
  ecore(c, T, T16, I, I16, jTI, TE, te, te16);
  ecore(c, I, I16, T, T16, jIT, IE, ie, ie16);

  float* tf  = (float*)c.ar.alloc((size_t)NTOK * DD * sizeof(float));
  float* imf = (float*)c.ar.alloc((size_t)NTOK * DD * sizeof(float));
  {
    c.ar.push();
    float* tc2 = (float*)c.ar.alloc((size_t)NTOK * DD * sizeof(float));
    float* ic2 = (float*)c.ar.alloc((size_t)NTOK * DD * sizeof(float));
    cma(c, te16, te, ie16, CM_BI, tc2, nullptr);
    cma(c, ie16, ie, te16, CM_BI, ic2, nullptr);
    h16* cat = (h16*)c.ar.alloc((size_t)NTOK * 2 * DD * sizeof(h16));
    h16* fh  = (h16*)c.ar.alloc((size_t)NTOK * 2 * DD * sizeof(h16));
    cvt_into(c, te, cat, NTOK, DD, 2 * DD, 0);
    cvt_into(c, tc2, cat, NTOK, DD, 2 * DD, DD);
    linear(c, cat, NTOK, 2 * DD, F1W, F1B, 2 * DD, nullptr, fh, ACT_GELU);
    linear(c, fh, NTOK, 2 * DD, F2W, F2B, DD, tf, nullptr, ACT_NONE);
    cvt_into(c, ie, cat, NTOK, DD, 2 * DD, 0);
    cvt_into(c, ic2, cat, NTOK, DD, 2 * DD, DD);
    linear(c, cat, NTOK, 2 * DD, F1W, F1B, 2 * DD, nullptr, fh, ACT_GELU);
    linear(c, fh, NTOK, 2 * DD, F2W, F2B, DD, imf, nullptr, ACT_NONE);
    c.ar.pop();
  }

  // ---- scale interaction (shared si weights, applied twice) ----
  {
    c.ar.push();
    h16* siA = (h16*)c.ar.alloc((size_t)NTOK * 3 * DD * sizeof(h16));
    cvt_into(c, local,  siA, NTOK, DD, 3 * DD, 0);
    cvt_into(c, out_sg, siA, NTOK, DD, 3 * DD, DD);
    cvt_into(c, tf,     siA, NTOK, DD, 3 * DD, 2 * DD);
    h16* WTsi = cvtW(c, SIW, 3 * DD, 3 * DD);
    gemm(c, siA, WTsi, c.P(SIB), nullptr, out_ft, nullptr, NTOK, 3 * DD, 3 * DD,
         3 * DD, 3 * DD, 3 * DD, 0, 1.0f, ACT_GELU);
    // stream order guarantees the first gemm reads siA before this rewrite executes
    cvt_into(c, imf, siA, NTOK, DD, 3 * DD, 2 * DD);
    gemm(c, siA, WTsi, c.P(SIB), nullptr, out_fi, nullptr, NTOK, 3 * DD, 3 * DD,
         3 * DD, 3 * DD, 3 * DD, 0, 1.0f, ACT_GELU);
    c.ar.pop();
  }
}
